// SubGraphMapping_45251775430768
// MI455X (gfx1250) — matrix operand order repaired and checked by reading, not compiled, not run on hardware
//
#include <hip/hip_runtime.h>
#include <hip/hip_bf16.h>
#include <math.h>

#define N_NODES 4096
#define BATCH   1024
#define NSUB    4
#define KSEL    64

typedef __attribute__((ext_vector_type(16))) _Float16 v16h;
typedef __attribute__((ext_vector_type(8)))  _Float16 v8h;
typedef __attribute__((ext_vector_type(8)))  float    v8f;

union V16U { v16h v; v8h h[2]; };

// ---------------------------------------------------------------------------
// Selection helpers: ordered (score,index) packing + wave32 shuffle reductions
// ---------------------------------------------------------------------------
__device__ __forceinline__ unsigned long long packKey(float v, int idx) {
  unsigned b = __float_as_uint(v);
  unsigned key = (b & 0x80000000u) ? ~b : (b | 0x80000000u);  // order-preserving
  return ((unsigned long long)key << 32) | (unsigned)(~idx);  // ties -> lower idx
}

__device__ __forceinline__ int keyIndex(unsigned long long k) {
  return (int)(~(unsigned)(k & 0xFFFFFFFFu));
}

__device__ __forceinline__ unsigned long long waveMax64(unsigned long long v) {
#pragma unroll
  for (int off = 16; off > 0; off >>= 1) {
    unsigned long long o = __shfl_xor(v, off, 32);
    v = (o > v) ? o : v;
  }
  return v;
}

// 1024-thread block max: wave shuffle + 32-leader LDS pass (3 barriers total)
__device__ __forceinline__ unsigned long long blockMax64(
    unsigned long long v, unsigned long long* red, int tid) {
  v = waveMax64(v);
  if ((tid & 31) == 0) red[tid >> 5] = v;
  __syncthreads();
  if (tid < 32) {
    unsigned long long w = waveMax64(red[tid]);
    if (tid == 0) red[0] = w;
  }
  __syncthreads();
  unsigned long long r = red[0];
  __syncthreads();   // protect red[] before next round's writes
  return r;
}

// ---------------------------------------------------------------------------
// Kernel 1: subgraph selection (top-4 degree, Gumbel top-64 w/o replacement)
// ---------------------------------------------------------------------------
__device__ __forceinline__ float urand01(unsigned i, unsigned n) {
  unsigned h = n * 0x9E3779B1u ^ (i + 1u) * 0x85EBCA6Bu;
  h ^= h >> 16; h *= 0x7FEB352Du;
  h ^= h >> 15; h *= 0x846CA68Bu;
  h ^= h >> 16;
  float u = (float)(h >> 8) * (1.0f / 16777216.0f);
  return fminf(fmaxf(u, 1e-7f), 0.99999994f);
}

__global__ __launch_bounds__(1024) void subgraph_select_kernel(
    const float* __restrict__ att, const float* __restrict__ deg,
    int* __restrict__ idx_out) {
  __shared__ unsigned long long sc64[N_NODES];   // packed (score,idx), 32KB
  __shared__ unsigned long long red[32];
  __shared__ unsigned selmask[N_NODES / 32];
  __shared__ int top4[NSUB];
  __shared__ int picks[KSEL];
  const int tid = threadIdx.x;

  for (int w = tid; w < N_NODES / 32; w += 1024) selmask[w] = 0u;
  if (tid < NSUB) top4[tid] = -1;

  // ---- top-4 degrees: pack once, 4 max passes clearing the winner ----
  for (int n = tid; n < N_NODES; n += 1024) sc64[n] = packKey(deg[n], n);
  __syncthreads();
  for (int t = 0; t < NSUB; ++t) {
    unsigned long long loc = 0ull;
    for (int n = tid; n < N_NODES; n += 1024) {
      unsigned long long v = sc64[n];
      loc = (v > loc) ? v : loc;
    }
    unsigned long long k = blockMax64(loc, red, tid);
    if (tid == 0) {
      int j = keyIndex(k);
      top4[t] = j;
      sc64[j] = 0ull;
    }
    __syncthreads();
  }

  // ---- per-subgraph Gumbel top-64 with global exclusion ----
  for (int i = 0; i < NSUB; ++i) {
    const int row = top4[i];
    for (int n = tid; n < N_NODES; n += 1024) {
      bool sel = (selmask[n >> 5] >> (n & 31)) & 1u;
      float p  = 200.0f * att[row * N_NODES + n];       // (att+att)*100
      float lp = __logf(p + 1e-12f);
      float u  = urand01((unsigned)i, (unsigned)n);
      float g  = -__logf(-__logf(u));                   // Gumbel noise
      sc64[n] = sel ? 0ull : packKey(lp + g, n);
    }
    __syncthreads();

    for (int p = 0; p < KSEL; ++p) {
      unsigned long long loc = 0ull;
      for (int n = tid; n < N_NODES; n += 1024) {
        unsigned long long v = sc64[n];
        loc = (v > loc) ? v : loc;
      }
      unsigned long long k = blockMax64(loc, red, tid);
      if (tid == 0) {
        int j = keyIndex(k);
        picks[p] = j;
        sc64[j] = 0ull;
        selmask[j >> 5] |= (1u << (j & 31));
      }
      __syncthreads();
    }

    if (tid == 0) {  // ascending sort (matches jnp.sort(idx))
      for (int a = 1; a < KSEL; ++a) {
        int key = picks[a]; int b = a - 1;
        while (b >= 0 && picks[b] > key) { picks[b + 1] = picks[b]; --b; }
        picks[b + 1] = key;
      }
    }
    __syncthreads();
    if (tid < KSEL) idx_out[i * KSEL + tid] = picks[tid];
    __syncthreads();
  }
}

// ---------------------------------------------------------------------------
// Kernel 2: convert W1, W2, Wa (each 64x64 f32) to f16 in workspace
// ---------------------------------------------------------------------------
__global__ __launch_bounds__(256) void wcvt_kernel(
    const float* __restrict__ a, const float* __restrict__ b,
    const float* __restrict__ c, _Float16* __restrict__ out) {
  int id = blockIdx.x * 256 + threadIdx.x;          // 0..12287
  const float* src = (id < 4096) ? a : (id < 8192 ? b : c);
  out[id] = (_Float16)src[id & 4095];
}

// ---------------------------------------------------------------------------
// Kernel 3: x_site = tanh(tanh(conv1d_same(x, w, b)))  (bandwidth-bound,
// float4 vectorized: one b128 load/store per 4 outputs)
// ---------------------------------------------------------------------------
__global__ __launch_bounds__(256) void site_conv_kernel(
    const float* __restrict__ x, const float* __restrict__ w,
    const float* __restrict__ bias, float* __restrict__ out) {
  int g4   = blockIdx.x * 256 + threadIdx.x;        // quad id
  int base = g4 * 4;
  int l    = base & (N_NODES - 1);                  // 0..4092 step 4
  float4 xc   = *(const float4*)(x + base);
  float left  = (l > 0)            ? x[base - 1] : 0.0f;
  float right = (l < N_NODES - 4)  ? x[base + 4] : 0.0f;
  const float w0 = w[0], w1 = w[1], w2 = w[2], b0 = bias[0];
  float4 y;
  y.x = fmaf(w0, left, fmaf(w1, xc.x, fmaf(w2, xc.y,  b0)));
  y.y = fmaf(w0, xc.x, fmaf(w1, xc.y, fmaf(w2, xc.z,  b0)));
  y.z = fmaf(w0, xc.y, fmaf(w1, xc.z, fmaf(w2, xc.w,  b0)));
  y.w = fmaf(w0, xc.z, fmaf(w1, xc.w, fmaf(w2, right, b0)));
  y.x = tanhf(tanhf(y.x));
  y.y = tanhf(tanhf(y.y));
  y.z = tanhf(tanhf(y.z));
  y.w = tanhf(tanhf(y.w));
  *(float4*)(out + base) = y;
}

// ---------------------------------------------------------------------------
// Kernel 4: fused gather + IMILmask (3 GEMMs via WMMA) + conv2 + double tanh
// Block = 4 batches -> 16-row tile (4 batches x 4 subgraphs), 4 waves, each
// wave owns one 16-wide N-tile of the 64 output features.
// ---------------------------------------------------------------------------
__global__ __launch_bounds__(128) void bag_kernel(
    const float* __restrict__ x, const int* __restrict__ idx,
    const _Float16* __restrict__ W1h, const _Float16* __restrict__ W2h,
    const _Float16* __restrict__ Wah,
    const float* __restrict__ bias1, const float* __restrict__ bias2,
    const float* __restrict__ biasa,
    const float* __restrict__ c2w, const float* __restrict__ c2b,
    float* __restrict__ out_bag) {
  __shared__ __align__(32) _Float16 Ah[16 * 64];   // mx tile, f16 (A operand)
  __shared__ __align__(32) float    Af[16 * 64];   // mx tile, f32 (for (A+1)*mx)
  __shared__ __align__(32) _Float16 L12[16 * 64];  // relu(l1)+relu(l2), f16
  __shared__ __align__(32) float    MXM[16 * 64];  // (sigmoid+1)*mx
  __shared__ int sidx[NSUB * KSEL];

  const int tid  = threadIdx.x;
  const int lane = tid & 31;
  const int wave = tid >> 5;          // N-tile id 0..3
  const int b0   = blockIdx.x * 4;

  if (tid < 128) { sidx[tid] = idx[tid]; sidx[tid + 128] = idx[tid + 128]; }
  __syncthreads();

  // gather mx: row r = local_batch*4 + subgraph, col = sampled node slot
#pragma unroll
  for (int q = 0; q < 8; ++q) {
    int e = tid + q * 128;
    int r = e >> 6, c = e & 63;
    int b = b0 + (r >> 2), j = r & 3;
    float v = x[b * N_NODES + sidx[j * KSEL + c]];
    Af[e] = v;
    Ah[e] = (_Float16)v;
  }
  __syncthreads();

  // Layout constants, contiguous K order k = 16*half + s (16-bit WMMA, wave32)
  const int n    = lane & 15;
  const int ng   = wave * 16 + n;        // global feature column 0..63
  const int kB   = (lane < 16) ? 0 : 16; // B operand: K block per lane half
  const int kA   = (lane < 16) ? 0 : 16; // A operand: K offset per lane half
  const int rowA = lane & 15;
  const int mOff = (lane < 16) ? 0 : 8;  // C/D operand: row offset per half

  __builtin_prefetch(&Wah[ng * 64], 0, 3);

  // A operand (mx) for both K-steps: elems 0..15 -> K kA..kA+15 (contiguous)
  V16U a0, a1;
  a0.h[0] = *(const v8h*)&Ah[rowA * 64 +  0 + kA];
  a0.h[1] = *(const v8h*)&Ah[rowA * 64 +  8 + kA];
  a1.h[0] = *(const v8h*)&Ah[rowA * 64 + 32 + kA];
  a1.h[1] = *(const v8h*)&Ah[rowA * 64 + 40 + kA];

  // B operand: B[k][n] = W[n][k] -> contiguous 16-halfword row slice of W
  const v16h bw10 = *(const v16h*)&W1h[ng * 64 +  0 + kB];
  const v16h bw11 = *(const v16h*)&W1h[ng * 64 + 32 + kB];
  const v16h bw20 = *(const v16h*)&W2h[ng * 64 +  0 + kB];
  const v16h bw21 = *(const v16h*)&W2h[ng * 64 + 32 + kB];

  v8f acc1 = {};
  acc1 = __builtin_amdgcn_wmma_f32_16x16x32_f16(false, a0.v, false, bw10, (short)0, acc1, false, false);
  acc1 = __builtin_amdgcn_wmma_f32_16x16x32_f16(false, a1.v, false, bw11, (short)0, acc1, false, false);
  v8f acc2 = {};
  acc2 = __builtin_amdgcn_wmma_f32_16x16x32_f16(false, a0.v, false, bw20, (short)0, acc2, false, false);
  acc2 = __builtin_amdgcn_wmma_f32_16x16x32_f16(false, a1.v, false, bw21, (short)0, acc2, false, false);

  const float bv1 = bias1[ng], bv2 = bias2[ng];
#pragma unroll
  for (int v = 0; v < 8; ++v) {
    float l = fmaxf(acc1[v] + bv1, 0.0f) + fmaxf(acc2[v] + bv2, 0.0f);
    L12[(v + mOff) * 64 + ng] = (_Float16)l;
  }
  __syncthreads();

  // second GEMM: (l1+l2) @ Wa.T
  V16U c0, c1;
  c0.h[0] = *(const v8h*)&L12[rowA * 64 +  0 + kA];
  c0.h[1] = *(const v8h*)&L12[rowA * 64 +  8 + kA];
  c1.h[0] = *(const v8h*)&L12[rowA * 64 + 32 + kA];
  c1.h[1] = *(const v8h*)&L12[rowA * 64 + 40 + kA];
  const v16h bwa0 = *(const v16h*)&Wah[ng * 64 +  0 + kB];
  const v16h bwa1 = *(const v16h*)&Wah[ng * 64 + 32 + kB];

  v8f acca = {};
  acca = __builtin_amdgcn_wmma_f32_16x16x32_f16(false, c0.v, false, bwa0, (short)0, acca, false, false);
  acca = __builtin_amdgcn_wmma_f32_16x16x32_f16(false, c1.v, false, bwa1, (short)0, acca, false, false);

  const float bva = biasa[ng];
#pragma unroll
  for (int v = 0; v < 8; ++v) {
    int m = v + mOff;
    float s = 1.0f / (1.0f + __expf(-(acca[v] + bva)));   // sigmoid
    MXM[m * 64 + ng] = (s + 1.0f) * Af[m * 64 + ng];      // (A_imi + 1) * mx
  }
  __syncthreads();

  // conv2 (4,4,3) SAME over channels + double tanh; 1024 outputs / block
#pragma unroll
  for (int q = 0; q < 8; ++q) {
    int e = tid + q * 128;
    int bl = e >> 8, o = (e >> 6) & 3, l = e & 63;
    float acc = c2b[o];
#pragma unroll
    for (int i = 0; i < 4; ++i) {
      int row = bl * 4 + i;
      float xm = (l > 0)  ? MXM[row * 64 + l - 1] : 0.0f;
      float xc = MXM[row * 64 + l];
      float xp = (l < 63) ? MXM[row * 64 + l + 1] : 0.0f;
      const float* wv = &c2w[(o * 4 + i) * 3];
      acc = fmaf(wv[0], xm, fmaf(wv[1], xc, fmaf(wv[2], xp, acc)));
    }
    out_bag[(b0 + bl) * 256 + o * 64 + l] = tanhf(tanhf(acc));
  }
}

// ---------------------------------------------------------------------------
extern "C" void kernel_launch(void* const* d_in, const int* in_sizes, int n_in,
                              void* d_out, int out_size, void* d_ws, size_t ws_size,
                              hipStream_t stream) {
  (void)in_sizes; (void)n_in; (void)out_size; (void)ws_size;

  const float* att = (const float*)d_in[0];
  const float* deg = (const float*)d_in[1];
  const float* x   = (const float*)d_in[2];
  const float* c1w = (const float*)d_in[3];
  const float* c1b = (const float*)d_in[4];
  const float* W1  = (const float*)d_in[5];
  const float* b1  = (const float*)d_in[6];
  const float* W2  = (const float*)d_in[7];
  const float* b2  = (const float*)d_in[8];
  const float* Wa  = (const float*)d_in[9];
  const float* ba  = (const float*)d_in[10];
  const float* c2w = (const float*)d_in[11];
  const float* c2b = (const float*)d_in[12];

  float* out    = (float*)d_out;
  float* x_bag  = out;                           // (1024, 4, 64)
  float* x_site = out + BATCH * NSUB * KSEL;     // (1024, 1, 4096)

  int*      idx_ws = (int*)d_ws;                          // 256 ints
  _Float16* Wh     = (_Float16*)((char*)d_ws + 1024);     // 3 x 4096 f16

  subgraph_select_kernel<<<1, 1024, 0, stream>>>(att, deg, idx_ws);
  wcvt_kernel<<<48, 256, 0, stream>>>(W1, W2, Wa, Wh);
  site_conv_kernel<<<(BATCH * N_NODES / 4) / 256, 256, 0, stream>>>(x, c1w, c1b, x_site);
  bag_kernel<<<BATCH / 4, 128, 0, stream>>>(x, idx_ws, Wh, Wh + 4096, Wh + 8192,
                                            b1, b2, ba, c2w, c2b, x_bag);
}
